// DeformableConv2d_35820027248864
// MI455X (gfx1250) — matrix operand order repaired and checked by reading, compiled, not run on hardware
//
#include <hip/hip_runtime.h>
#include <stdint.h>

// ---------------- types ----------------
typedef __attribute__((ext_vector_type(16))) __bf16 v16bf;
typedef __attribute__((ext_vector_type(8)))  float  v8f;
typedef __attribute__((ext_vector_type(4)))  int    v4i;

// problem constants (from reference setup_inputs)
#define BATCH 4
#define CIN   128
#define COUT  256
#define HH    64
#define WW    64
#define KK    9            // 3x3 taps
#define MPB   (HH*WW)      // 4096 output positions per batch
#define KTOT  (KK*CIN)     // 1152 contraction length (tap-major: k = tap*CIN + c)

// GEMM tiling
#define BM 128
#define BN 128
#define BK 32

static __device__ __forceinline__ unsigned short f2bf(float f) {
  __bf16 h = (__bf16)f;                    // native v_cvt path on gfx1250
  union { __bf16 b; unsigned short s; } u;
  u.b = h;
  return u.s;
}

// ---------------- prep kernels ----------------

// NCHW -> NHWC (float): xT[(b*4096+hw)*128 + c]
__global__ __launch_bounds__(256) void prep_transpose(const float* __restrict__ x,
                                                      float* __restrict__ xT) {
  int t = blockIdx.x * 256 + threadIdx.x;        // 2,097,152 total
  int c = t & (CIN - 1);
  int s = t >> 7;                                 // b*4096 + hw
  int b = s >> 12;
  int hw = s & (MPB - 1);
  xT[(size_t)s * CIN + c] = x[((size_t)(b * CIN + c)) * MPB + hw];
}

// weight [Cout][Cin][3][3] f32 -> wB2[cout][tap*128+c] bf16
__global__ __launch_bounds__(256) void prep_packW(const float* __restrict__ w,
                                                  unsigned short* __restrict__ wB2) {
  int t = blockIdx.x * 256 + threadIdx.x;        // 294,912 total
  int cout = t / KTOT;
  int kg = t - cout * KTOT;
  int tap = kg >> 7;
  int c = kg & (CIN - 1);
  wB2[(size_t)cout * KTOT + kg] = f2bf(w[((size_t)(cout * CIN + c)) * KK + tap]);
}

// offset_w[18][Cin][3][3] + mod_w[9][Cin][3][3] -> wOM[(tap*128+c)*27 + ch]
__global__ __launch_bounds__(256) void prep_packOM(const float* __restrict__ ow,
                                                   const float* __restrict__ mw,
                                                   float* __restrict__ wOM) {
  int t = blockIdx.x * 256 + threadIdx.x;
  if (t >= KTOT * 27) return;
  int ch = t % 27;
  int rest = t / 27;                              // tap*128 + c
  int tap = rest >> 7;
  int c = rest & (CIN - 1);
  float v = (ch < 18) ? ow[((size_t)(ch * CIN + c)) * KK + tap]
                      : mw[((size_t)((ch - 18) * CIN + c)) * KK + tap];
  wOM[t] = v;
}

// ---------------- offset / modulation conv ----------------
__global__ __launch_bounds__(256) void offmask_conv(const float* __restrict__ xT,
                                                    const float* __restrict__ wOM,
                                                    const float* __restrict__ off_b,
                                                    const float* __restrict__ mod_b,
                                                    float* __restrict__ offs,
                                                    float* __restrict__ maskA) {
  int g = blockIdx.x * 256 + threadIdx.x;         // 16384 total
  int b = g >> 12;
  int m = g & (MPB - 1);
  int hh = m >> 6, ww = m & 63;

  float acc[27];
#pragma unroll
  for (int i = 0; i < 27; ++i) acc[i] = 0.0f;

  for (int ky = 0; ky < 3; ++ky) {
    int y = hh - 1 + ky;
    if (y < 0 || y >= HH) continue;
    for (int kx = 0; kx < 3; ++kx) {
      int x = ww - 1 + kx;
      if (x < 0 || x >= WW) continue;
      int tap = ky * 3 + kx;
      const float* xp = xT + ((size_t)(b * MPB + y * WW + x)) * CIN;
      const float* wp = wOM + (size_t)tap * CIN * 27;
      for (int c = 0; c < CIN; ++c) {
        float xv = xp[c];
        const float* wr = wp + c * 27;            // wave-uniform address -> scalar path
#pragma unroll
        for (int ch = 0; ch < 27; ++ch) acc[ch] = fmaf(xv, wr[ch], acc[ch]);
      }
    }
  }

  size_t base = (size_t)(b * MPB + m) * KK;
#pragma unroll
  for (int kk = 0; kk < KK; ++kk) {
    float oy = fminf(fmaxf(acc[kk * 2 + 0] + off_b[kk * 2 + 0], -16.0f), 16.0f);
    float ox = fminf(fmaxf(acc[kk * 2 + 1] + off_b[kk * 2 + 1], -16.0f), 16.0f);
    float mv = 2.0f / (1.0f + __expf(-(acc[18 + kk] + mod_b[kk])));
    offs[(base + kk) * 2 + 0] = oy;
    offs[(base + kk) * 2 + 1] = ox;
    maskA[base + kk] = mv;
  }
}

// ---------------- fused sampling + bf16 WMMA GEMM ----------------
__global__ __launch_bounds__(256) void dcn_gemm(const float* __restrict__ xT,
                                                const unsigned short* __restrict__ wB2,
                                                const float* __restrict__ offs,
                                                const float* __restrict__ maskA,
                                                float* __restrict__ out) {
  __shared__ alignas(16) unsigned short As[BM][BK];   // [row][k-local] 8KB
  __shared__ alignas(16) unsigned short Bs[BN][BK];   // [n-local][k-local] 8KB

  const int tid = threadIdx.x;
  const int m0 = blockIdx.x * BM;                 // within batch
  const int n0 = blockIdx.y * BN;
  const int b  = blockIdx.z;

  // sampler role
  const int row  = tid & (BM - 1);
  const int half = tid >> 7;                       // 0/1: which 16 channels of the 32-chunk
  const int m  = m0 + row;
  const int hh = m >> 6, ww = m & 63;

  // wmma role: 8 waves as 2(M) x 4(N); wave tile 64x32 -> 4x2 wmma frags
  const int lane = tid & 31;
  const int wid  = tid >> 5;
  const int waveM = wid & 1;
  const int waveN = wid >> 1;
  const int lhi = lane >> 4;                       // lane half
  const int llo = lane & 15;
  const int kbA = lhi * 8;                         // A frag: K chunks [kbA..kbA+7],[kbA+16..kbA+23]
  const int kbB = lhi * 8;                         // B frag: K chunks [kbB..kbB+7],[kbB+16..kbB+23]

  v8f acc[4][2];
  const v8f zeroV = {0.f, 0.f, 0.f, 0.f, 0.f, 0.f, 0.f, 0.f};
#pragma unroll
  for (int i = 0; i < 4; ++i)
#pragma unroll
    for (int j = 0; j < 2; ++j) acc[i][j] = zeroV;

  for (int tap = 0; tap < KK; ++tap) {
    // ---- per-(row, tap) sampling parameters (shared over all 128 channels) ----
    const size_t ob = (size_t)(b * MPB + m) * KK + tap;
    const float offy = offs[ob * 2 + 0];
    const float offx = offs[ob * 2 + 1];
    const float mval = maskA[ob];
    const int ty = tap / 3, tx = tap - ty * 3;
    const float py = (float)(hh - 1 + ty) + offy;
    const float px = (float)(ww - 1 + tx) + offx;
    const float y0f = floorf(py), x0f = floorf(px);
    const float fy = py - y0f, fx = px - x0f;
    const int iy0 = (int)y0f, ix0 = (int)x0f;
    const int iy1 = iy0 + 1,  ix1 = ix0 + 1;
    const float vy0 = (iy0 >= 0 && iy0 < HH) ? 1.0f : 0.0f;
    const float vy1 = (iy1 >= 0 && iy1 < HH) ? 1.0f : 0.0f;
    const float vx0 = (ix0 >= 0 && ix0 < WW) ? 1.0f : 0.0f;
    const float vx1 = (ix1 >= 0 && ix1 < WW) ? 1.0f : 0.0f;
    const float w00 = (1.f - fy) * (1.f - fx) * mval * vy0 * vx0;
    const float w01 = (1.f - fy) * fx         * mval * vy0 * vx1;
    const float w10 = fy * (1.f - fx)         * mval * vy1 * vx0;
    const float w11 = fy * fx                 * mval * vy1 * vx1;
    const int cy0 = min(max(iy0, 0), HH - 1), cy1 = min(max(iy1, 0), HH - 1);
    const int cx0 = min(max(ix0, 0), WW - 1), cx1 = min(max(ix1, 0), WW - 1);
    const float* p00 = xT + ((size_t)(b * MPB + cy0 * WW + cx0)) * CIN;
    const float* p01 = xT + ((size_t)(b * MPB + cy0 * WW + cx1)) * CIN;
    const float* p10 = xT + ((size_t)(b * MPB + cy1 * WW + cx0)) * CIN;
    const float* p11 = xT + ((size_t)(b * MPB + cy1 * WW + cx1)) * CIN;

    for (int cc = 0; cc < 4; ++cc) {              // Cin chunks of 32
      __syncthreads();                            // previous compute done reading LDS

      // ---- B tile: async copy global -> LDS (ASYNCcnt) when available ----
      const unsigned short* bw =
          wB2 + (size_t)(n0 + row) * KTOT + tap * CIN + cc * 32 + half * 16;
#if __has_builtin(__builtin_amdgcn_global_load_async_to_lds_b128)
      {
        // signature per hipcc diagnostics: (AS1 v4i*, AS3 v4i*, imm off, imm cpol)
        __attribute__((address_space(1))) v4i* gp =
            (__attribute__((address_space(1))) v4i*)(uintptr_t)bw;
        __attribute__((address_space(3))) v4i* lp =
            (__attribute__((address_space(3))) v4i*)&Bs[row][half * 16];
        __builtin_amdgcn_global_load_async_to_lds_b128(gp, lp, 0, 0);
        __builtin_amdgcn_global_load_async_to_lds_b128(gp, lp, 16, 0);
      }
#else
      *(float4*)&Bs[row][half * 16 + 0] = *(const float4*)(bw);
      *(float4*)&Bs[row][half * 16 + 8] = *(const float4*)(bw + 8);
#endif

      // ---- build A tile: 2 threads/row, 16 channels each ----
      const int c0 = cc * 32 + half * 16;
      __builtin_prefetch(p00 + c0 + 32, 0, 3);    // gfx1250 global_prefetch_b8
#pragma unroll
      for (int j = 0; j < 4; ++j) {
        const float4 v00 = *(const float4*)(p00 + c0 + j * 4);
        const float4 v01 = *(const float4*)(p01 + c0 + j * 4);
        const float4 v10 = *(const float4*)(p10 + c0 + j * 4);
        const float4 v11 = *(const float4*)(p11 + c0 + j * 4);
        union { __bf16 h[4]; unsigned int u[2]; } pk;
        pk.h[0] = (__bf16)(w00 * v00.x + w01 * v01.x + w10 * v10.x + w11 * v11.x);
        pk.h[1] = (__bf16)(w00 * v00.y + w01 * v01.y + w10 * v10.y + w11 * v11.y);
        pk.h[2] = (__bf16)(w00 * v00.z + w01 * v01.z + w10 * v10.z + w11 * v11.z);
        pk.h[3] = (__bf16)(w00 * v00.w + w01 * v01.w + w10 * v10.w + w11 * v11.w);
        *(unsigned int*)&As[row][half * 16 + j * 4 + 0] = pk.u[0];
        *(unsigned int*)&As[row][half * 16 + j * 4 + 2] = pk.u[1];
      }

#if __has_builtin(__builtin_amdgcn_global_load_async_to_lds_b128)
#if __has_builtin(__builtin_amdgcn_s_wait_asynccnt)
      __builtin_amdgcn_s_wait_asynccnt(0);
#else
      asm volatile("s_wait_asynccnt 0x0" ::: "memory");
#endif
#endif
      __syncthreads();                            // tiles ready

      // ---- WMMA compute: 4x2 fragments of 16x16, K=32 bf16 ----
      v16bf bfrag[2];
#pragma unroll
      for (int ns = 0; ns < 2; ++ns) {
        const int n = waveN * 32 + ns * 16 + llo;
        const unsigned short* bp = &Bs[n][kbB];   // B: chunks [kbB..+7] and [kbB+16..+23]
        ((float4*)&bfrag[ns])[0] = *(const float4*)(bp);
        ((float4*)&bfrag[ns])[1] = *(const float4*)(bp + 16);
      }
#pragma unroll
      for (int ms = 0; ms < 4; ++ms) {
        const int r = waveM * 64 + ms * 16 + llo;
        v16bf afrag;                              // A: chunks [kbA..+7] and [kbA+16..+23]
        ((float4*)&afrag)[0] = *(const float4*)(&As[r][kbA]);
        ((float4*)&afrag)[1] = *(const float4*)(&As[r][kbA + 16]);
#pragma unroll
        for (int ns = 0; ns < 2; ++ns) {
          acc[ms][ns] = __builtin_amdgcn_wmma_f32_16x16x32_bf16(
              false, afrag, false, bfrag[ns], (short)0, acc[ms][ns], false, false);
        }
      }
    }
  }

  // ---- epilogue: C/D layout — VGPR v: M = v + 8*lanehalf, N = lane&15 ----
#pragma unroll
  for (int ms = 0; ms < 4; ++ms) {
#pragma unroll
    for (int ns = 0; ns < 2; ++ns) {
      const int n = n0 + waveN * 32 + ns * 16 + llo;
      float* op = out + ((size_t)(b * COUT + n)) * MPB
                + m0 + waveM * 64 + ms * 16 + 8 * lhi;
#pragma unroll
      for (int v = 0; v < 8; ++v) op[v] = acc[ms][ns][v];
    }
  }
}

// ---------------- launch ----------------
extern "C" void kernel_launch(void* const* d_in, const int* in_sizes, int n_in,
                              void* d_out, int out_size, void* d_ws, size_t ws_size,
                              hipStream_t stream) {
  const float* x        = (const float*)d_in[0];
  const float* offset_w = (const float*)d_in[1];
  const float* offset_b = (const float*)d_in[2];
  const float* mod_w    = (const float*)d_in[3];
  const float* mod_b    = (const float*)d_in[4];
  const float* weight   = (const float*)d_in[5];
  float* out = (float*)d_out;

  char* ws = (char*)d_ws;
  float*          xT    = (float*)(ws + 0);                     // 8,388,608 B
  unsigned short* wB2   = (unsigned short*)(ws + 8388608);      //   589,824 B
  float*          wOM   = (float*)(ws + 8978432);               //   124,416 B
  float*          offsA = (float*)(ws + 9102848);               // 1,179,648 B
  float*          maskA = (float*)(ws + 10282496);              //   589,824 B

  prep_transpose<<<8192, 256, 0, stream>>>(x, xT);
  prep_packW<<<1152, 256, 0, stream>>>(weight, wB2);
  prep_packOM<<<122, 256, 0, stream>>>(offset_w, mod_w, wOM);
  offmask_conv<<<64, 256, 0, stream>>>(xT, wOM, offset_b, mod_b, offsA, maskA);
  dcn_gemm<<<dim3(MPB / BM, COUT / BN, BATCH), 256, 0, stream>>>(
      xT, wB2, offsA, maskA, out);
}
